// MoE_67619965108995
// MI455X (gfx1250) — matrix operand order repaired and checked by reading, not compiled, not run on hardware
//
#include <hip/hip_runtime.h>

// ---------------- types ----------------
typedef __attribute__((ext_vector_type(16))) __bf16 v16bf;
typedef __attribute__((ext_vector_type(8)))  __bf16 bf16x8;
typedef __attribute__((ext_vector_type(4)))  __bf16 bf16x4;
typedef __attribute__((ext_vector_type(8)))  float  v8f;
typedef __attribute__((ext_vector_type(4)))  float  f32x4;
typedef __attribute__((ext_vector_type(4)))  int    v4i;

#define GLOBAL_AS __attribute__((address_space(1)))
#define LDS_AS    __attribute__((address_space(3)))

// ---------------- problem constants ----------------
constexpr int NE   = 16;       // experts
constexpr int TOPK = 2;
constexpr int NB   = 8;        // batch
constexpr int NT   = 4096;     // tokens per batch
constexpr int NC   = 1024;     // model dim
constexpr int NH   = 512;      // hidden dim
constexpr int CAP  = 512;      // per-batch expert capacity = K*T/E
constexpr int NROW = NB * CAP; // 4096 rows per expert buffer

// Build a 16-element bf16 WMMA fragment from two contiguous 16-byte chunks.
// A layout used here (contiguous K, same as B): lane half h holds K = {kk+16h .. +15}.
// B layout: lane half h holds K = {kk+16h .. +15}, so callers pass (p, p+8).
static __device__ inline v16bf frag16(const __bf16* p0, const __bf16* p1) {
    bf16x8 lo = *(const bf16x8*)p0;
    bf16x8 hi = *(const bf16x8*)p1;
    v16bf r;
#pragma unroll
    for (int i = 0; i < 8; ++i) { r[i] = lo[i]; r[i + 8] = hi[i]; }
    return r;
}

// 16B global -> LDS async copy (ASYNCcnt path); falls back to a plain copy.
static __device__ inline void async_copy16(const __bf16* g, __bf16* l) {
#if __has_builtin(__builtin_amdgcn_global_load_async_to_lds_b128)
    __builtin_amdgcn_global_load_async_to_lds_b128(
        (GLOBAL_AS v4i*)(GLOBAL_AS void*)g,
        (LDS_AS v4i*)(LDS_AS void*)l, 0, 0);
#else
    *(bf16x8*)l = *(const bf16x8*)g;
#endif
}

static __device__ inline void async_wait0() {
#if __has_builtin(__builtin_amdgcn_global_load_async_to_lds_b128)
#if __has_builtin(__builtin_amdgcn_s_wait_asynccnt)
    __builtin_amdgcn_s_wait_asynccnt(0);
#else
    asm volatile("s_wait_asynccnt 0" ::: "memory");
#endif
#endif
}

// ---------------- 1: weight convert + transpose to bf16 ----------------
// Wg_t/Wf_t: [E][H][C]  (row = output hidden unit, contiguous in C=K)
// Wp_t:      [E][C][H]  (row = output channel,     contiguous in H=K)
__global__ void moe_convert_w(const float* __restrict__ w_gate,
                              const float* __restrict__ w_fc,
                              const float* __restrict__ w_proj,
                              __bf16* __restrict__ Wg,
                              __bf16* __restrict__ Wf,
                              __bf16* __restrict__ Wp) {
    size_t i = (size_t)blockIdx.x * blockDim.x + threadIdx.x;
    const size_t n = (size_t)NE * NH * NC;
    if (i >= n) return;
    // gate/fc: out [e][h][c] <- in [e][c][h]
    int c = (int)(i % NC);
    size_t t = i / NC;
    int h = (int)(t % NH);
    int e = (int)(t / NH);
    Wg[i] = (__bf16)w_gate[((size_t)e * NC + c) * NH + h];
    Wf[i] = (__bf16)w_fc  [((size_t)e * NC + c) * NH + h];
    // proj: out [e][co][h2] <- in [e][h2][co]
    int h2 = (int)(i % NH);
    size_t t2 = i / NH;
    int co = (int)(t2 % NC);
    int e2 = (int)(t2 / NC);
    Wp[i] = (__bf16)w_proj[((size_t)e2 * NH + h2) * NC + co];
}

// ---------------- 2: router (logits, softmax, top-2) ----------------
__global__ __launch_bounds__(256) void moe_router(const float* __restrict__ x,
                                                  const float* __restrict__ w_router,
                                                  int* __restrict__ idx,
                                                  float* __restrict__ probs) {
    const int tok  = blockIdx.x * 8 + (threadIdx.x >> 5);
    const int lane = threadIdx.x & 31;
    float acc[NE];
#pragma unroll
    for (int e = 0; e < NE; ++e) acc[e] = 0.0f;
    const float* xr = x + (size_t)tok * NC;
    for (int c = lane; c < NC; c += 32) {
        const float xv = xr[c];
        const float* w = w_router + (size_t)c * NE;
#pragma unroll
        for (int e = 0; e < NE; ++e) acc[e] += xv * w[e];
    }
#pragma unroll
    for (int m = 16; m >= 1; m >>= 1) {
#pragma unroll
        for (int e = 0; e < NE; ++e) acc[e] += __shfl_xor(acc[e], m, 32);
    }
    if (lane == 0) {
        float mx = acc[0];
#pragma unroll
        for (int e = 1; e < NE; ++e) mx = fmaxf(mx, acc[e]);
        float p[NE]; float s = 0.0f;
#pragma unroll
        for (int e = 0; e < NE; ++e) { p[e] = __expf(acc[e] - mx); s += p[e]; }
        const float inv = __builtin_amdgcn_rcpf(s);
        int b0 = 0; float v0 = -1.0f;
#pragma unroll
        for (int e = 0; e < NE; ++e) if (p[e] > v0) { v0 = p[e]; b0 = e; }
        int b1 = 0; float v1 = -1.0f;
#pragma unroll
        for (int e = 0; e < NE; ++e) if (e != b0 && p[e] > v1) { v1 = p[e]; b1 = e; }
        idx[(size_t)tok * 2 + 0] = b0;
        idx[(size_t)tok * 2 + 1] = b1;
        probs[(size_t)tok * 2 + 0] = v0 * inv;
        probs[(size_t)tok * 2 + 1] = v1 * inv;
    }
}

// ---------------- 3: per-(batch,expert) running positions, k-major ----------------
__global__ void moe_positions(const int* __restrict__ idx, int* __restrict__ pos) {
    const int b = blockIdx.x;
    const int lane = threadIdx.x; // 32 lanes, lanes 0..15 own expert == lane
    int cnt = 0;
    for (int k = 0; k < TOPK; ++k) {
        for (int t = 0; t < NT; ++t) {
            const int e = idx[((size_t)b * NT + t) * 2 + k];
            if (lane < NE && e == lane) {
                pos[((size_t)b * NT + t) * 2 + k] = cnt;
                ++cnt;
            }
        }
    }
}

// ---------------- 4: zero the bf16 expert-input buffer ----------------
__global__ void moe_zero(uint4* __restrict__ p, size_t n) {
    size_t i = (size_t)blockIdx.x * blockDim.x + threadIdx.x;
    const size_t stride = (size_t)gridDim.x * blockDim.x;
    uint4 z; z.x = z.y = z.z = z.w = 0u;
    for (; i < n; i += stride) p[i] = z;
}

// ---------------- 5: dispatch scatter f32 -> bf16 rows ----------------
__global__ __launch_bounds__(64) void moe_dispatch(const float* __restrict__ x,
                                                   const int* __restrict__ idx,
                                                   const int* __restrict__ pos,
                                                   __bf16* __restrict__ bufX) {
    const int g = blockIdx.x;              // == (b*NT + t)*2 + k
    const int p = pos[g];
    if (p >= CAP) return;                  // capacity overflow: dropped
    const int e = idx[g];
    const int bt = g >> 1;                 // b*NT + t
    const int b = bt >> 12;                // NT = 4096
    const float* src = x + (size_t)bt * NC;
    __bf16* dst = bufX + ((size_t)e * NROW + (size_t)b * CAP + p) * NC;
    for (int c = threadIdx.x * 4; c < NC; c += 64 * 4) {
        f32x4 v = *(const f32x4*)(src + c);
        bf16x4 o;
        o[0] = (__bf16)v.x; o[1] = (__bf16)v.y; o[2] = (__bf16)v.z; o[3] = (__bf16)v.w;
        *(bf16x4*)(dst + c) = o;
    }
}

// ---------------- 6: fused grouped SwiGLU expert GEMM ----------------
// Block = 128 threads (4 wave32). Each wave owns TWO 16-row A tiles (32 rows),
// so every B fragment feeds 2 WMMAs (halves B traffic per FLOP). Block = 128 rows.
// Stage 1: G = X*Wg^T, F = X*Wf^T (v_wmma_f32_16x16x32_bf16); H = silu(G)*F -> LDS bf16.
// Stage 2: O = H*Wp^T; Wp tile staged cooperatively via global_load_async_to_lds_b128.
constexpr int HSTRIDE = 520; // 512 + 8 bf16 pad, rows stay 16B aligned

__global__ __launch_bounds__(128) void moe_expert_gemm(const __bf16* __restrict__ bufX, // [E][NROW][C]
                                                       const __bf16* __restrict__ Wg,   // [E][H][C]
                                                       const __bf16* __restrict__ Wf,   // [E][H][C]
                                                       const __bf16* __restrict__ Wp,   // [E][C][H]
                                                       float* __restrict__ O) {         // [E][NROW][C]
    __shared__ alignas(16) __bf16 Hls[128 * HSTRIDE]; // 133 KB: H tile, 128 rows
    __shared__ alignas(16) __bf16 Bls[16 * HSTRIDE];  // 16.6 KB: staged Wp tile (16 cols x 512 K)

    const int e    = blockIdx.y;
    const int rb   = blockIdx.x * 128;
    const int wave = threadIdx.x >> 5;
    const int lane = threadIdx.x & 31;
    const int n16  = lane & 15;   // column / A-row within tile
    const int half = lane >> 4;   // K-half select

    const __bf16* Arow0 = bufX + ((size_t)e * NROW + rb + wave * 32 +      n16) * NC;
    const __bf16* Arow1 = bufX + ((size_t)e * NROW + rb + wave * 32 + 16 + n16) * NC;

    // ---- Stage 1: gate & fc GEMMs + SwiGLU, 64 H-columns at a time ----
#pragma unroll 1
    for (int hc = 0; hc < NH / 64; ++hc) {
        v8f accG[2][4], accF[2][4];
#pragma unroll
        for (int r = 0; r < 2; ++r)
#pragma unroll
            for (int t = 0; t < 4; ++t) { accG[r][t] = v8f{}; accF[r][t] = v8f{}; }

#pragma unroll 1
        for (int kk = 0; kk < NC; kk += 32) {
            const v16bf a0 = frag16(Arow0 + kk + 16 * half, Arow0 + kk + 16 * half + 8);
            const v16bf a1 = frag16(Arow1 + kk + 16 * half, Arow1 + kk + 16 * half + 8);
#pragma unroll
            for (int t = 0; t < 4; ++t) {
                const int nn = hc * 64 + t * 16 + n16;
                const __bf16* gp = Wg + ((size_t)e * NH + nn) * NC + kk + 16 * half;
                const __bf16* fp = Wf + ((size_t)e * NH + nn) * NC + kk + 16 * half;
                const v16bf bg = frag16(gp, gp + 8);
                const v16bf bf = frag16(fp, fp + 8);
                accG[0][t] = __builtin_amdgcn_wmma_f32_16x16x32_bf16(
                    false, a0, false, bg, (short)0, accG[0][t], false, false);
                accG[1][t] = __builtin_amdgcn_wmma_f32_16x16x32_bf16(
                    false, a1, false, bg, (short)0, accG[1][t], false, false);
                accF[0][t] = __builtin_amdgcn_wmma_f32_16x16x32_bf16(
                    false, a0, false, bf, (short)0, accF[0][t], false, false);
                accF[1][t] = __builtin_amdgcn_wmma_f32_16x16x32_bf16(
                    false, a1, false, bf, (short)0, accF[1][t], false, false);
            }
        }
        // SwiGLU elementwise in f32, park H tile in LDS as bf16
#pragma unroll
        for (int r = 0; r < 2; ++r) {
#pragma unroll
            for (int t = 0; t < 4; ++t) {
#pragma unroll
                for (int j = 0; j < 8; ++j) {
                    const float g  = accG[r][t][j];
                    const float f  = accF[r][t][j];
                    // silu(g)*f with hardware v_rcp_f32 (no IEEE div expansion)
                    const float hv = g * __builtin_amdgcn_rcpf(1.0f + __expf(-g)) * f;
                    const int rl = wave * 32 + r * 16 + j + 8 * half; // C-matrix row mapping
                    const int cc = hc * 64 + t * 16 + n16;
                    Hls[rl * HSTRIDE + cc] = (__bf16)hv;
                }
            }
        }
    }
    __syncthreads();

    // ---- Stage 2: projection O = H * Wp ----
    const __bf16* Hrow0 = Hls + (size_t)(wave * 32 +      n16) * HSTRIDE;
    const __bf16* Hrow1 = Hls + (size_t)(wave * 32 + 16 + n16) * HSTRIDE;
    float* Obase = O + ((size_t)e * NROW + rb + wave * 32) * NC;
#pragma unroll 1
    for (int ct = 0; ct < NC / 16; ++ct) {
        // Cooperative async stage of the 16 x 512 bf16 Wp tile (16 KB) into LDS.
        // 1024 x 16B chunks across 128 threads -> 8 async copies per thread.
#pragma unroll
        for (int i = 0; i < 8; ++i) {
            const int chunk = threadIdx.x + i * 128;
            const int row = chunk >> 6;   // 0..15  (64 x 16B chunks per 1KB row)
            const int c16 = chunk & 63;
            async_copy16(Wp + ((size_t)e * NC + ct * 16 + row) * NH + c16 * 8,
                         Bls + row * HSTRIDE + c16 * 8);
        }
        async_wait0();
        __syncthreads();

        // Prefetch the next tile's rows while we compute (global_prefetch_b8).
        if (ct + 1 < NC / 16 && lane < 16)
            __builtin_prefetch(Wp + ((size_t)e * NC + (ct + 1) * 16 + lane) * NH, 0, 3);

        v8f acc0 = v8f{}, acc1 = v8f{};
#pragma unroll 1
        for (int kk = 0; kk < NH; kk += 32) {
            const v16bf b  = frag16(Bls + n16 * HSTRIDE + kk + 16 * half,
                                    Bls + n16 * HSTRIDE + kk + 16 * half + 8);
            const v16bf a0 = frag16(Hrow0 + kk + 16 * half, Hrow0 + kk + 16 * half + 8);
            const v16bf a1 = frag16(Hrow1 + kk + 16 * half, Hrow1 + kk + 16 * half + 8);
            acc0 = __builtin_amdgcn_wmma_f32_16x16x32_bf16(
                false, a0, false, b, (short)0, acc0, false, false);
            acc1 = __builtin_amdgcn_wmma_f32_16x16x32_bf16(
                false, a1, false, b, (short)0, acc1, false, false);
        }
        __syncthreads(); // everyone done reading Bls before next stage overwrite

#pragma unroll
        for (int j = 0; j < 8; ++j) {
            Obase[(size_t)(     j + 8 * half) * NC + ct * 16 + n16] = acc0[j];
            Obase[(size_t)(16 + j + 8 * half) * NC + ct * 16 + n16] = acc1[j];
        }
    }
}

// ---------------- 7: collect (gather + weighted sum) ----------------
__global__ __launch_bounds__(128) void moe_collect(const float* __restrict__ O,
                                                   const int* __restrict__ idx,
                                                   const int* __restrict__ pos,
                                                   const float* __restrict__ probs,
                                                   float* __restrict__ y) {
    const int g = blockIdx.x;   // b*NT + t
    const int b = g >> 12;      // NT = 4096
    const int e0 = idx[(size_t)g * 2 + 0];
    const int e1 = idx[(size_t)g * 2 + 1];
    int p0 = pos[(size_t)g * 2 + 0]; p0 = p0 < CAP - 1 ? p0 : CAP - 1; // CLIP-gather like jnp
    int p1 = pos[(size_t)g * 2 + 1]; p1 = p1 < CAP - 1 ? p1 : CAP - 1;
    const float w0 = probs[(size_t)g * 2 + 0];
    const float w1 = probs[(size_t)g * 2 + 1];
    const float* r0 = O + ((size_t)e0 * NROW + (size_t)b * CAP + p0) * NC;
    const float* r1 = O + ((size_t)e1 * NROW + (size_t)b * CAP + p1) * NC;
    float* yr = y + (size_t)g * NC;
    for (int c = threadIdx.x * 4; c < NC; c += 128 * 4) {
        const f32x4 a  = *(const f32x4*)(r0 + c);
        const f32x4 bb = *(const f32x4*)(r1 + c);
        f32x4 r;
        r.x = w0 * a.x + w1 * bb.x;
        r.y = w0 * a.y + w1 * bb.y;
        r.z = w0 * a.z + w1 * bb.z;
        r.w = w0 * a.w + w1 * bb.w;
        *(f32x4*)(yr + c) = r;
    }
}

// ---------------- launch ----------------
extern "C" void kernel_launch(void* const* d_in, const int* in_sizes, int n_in,
                              void* d_out, int out_size, void* d_ws, size_t ws_size,
                              hipStream_t stream) {
    const float* x        = (const float*)d_in[0];
    const float* w_router = (const float*)d_in[1];
    const float* w_fc     = (const float*)d_in[2];
    const float* w_gate   = (const float*)d_in[3];
    const float* w_proj   = (const float*)d_in[4];
    float* y = (float*)d_out;

    char* ws = (char*)d_ws;
    size_t off = 0;
    auto take = [&](size_t bytes) -> char* {
        char* p = ws + off;
        off = (off + bytes + 255) & ~(size_t)255;
        return p;
    };
    __bf16* bufX  = (__bf16*)take((size_t)NE * NROW * NC * 2);   // 134 MB
    float*  Obuf  = (float*) take((size_t)NE * NROW * NC * 4);   // 268 MB
    __bf16* Wg    = (__bf16*)take((size_t)NE * NH * NC * 2);     // 16.8 MB
    __bf16* Wf    = (__bf16*)take((size_t)NE * NH * NC * 2);
    __bf16* Wp    = (__bf16*)take((size_t)NE * NC * NH * 2);
    int*    idx   = (int*)   take((size_t)NB * NT * TOPK * 4);
    int*    pos   = (int*)   take((size_t)NB * NT * TOPK * 4);
    float*  probs = (float*) take((size_t)NB * NT * TOPK * 4);
    (void)in_sizes; (void)n_in; (void)out_size; (void)ws_size;

    const size_t nw = (size_t)NE * NH * NC;
    moe_convert_w<<<(unsigned)((nw + 255) / 256), 256, 0, stream>>>(w_gate, w_fc, w_proj, Wg, Wf, Wp);
    moe_router<<<NB * NT / 8, 256, 0, stream>>>(x, w_router, idx, probs);
    moe_positions<<<NB, 32, 0, stream>>>(idx, pos);
    moe_zero<<<4096, 256, 0, stream>>>((uint4*)bufX, (size_t)NE * NROW * NC * 2 / 16);
    moe_dispatch<<<NB * NT * TOPK, 64, 0, stream>>>(x, idx, pos, bufX);
    dim3 gg(NROW / 128, NE);
    moe_expert_gemm<<<gg, 128, 0, stream>>>(bufX, Wg, Wf, Wp, Obuf);
    moe_collect<<<NB * NT, 128, 0, stream>>>(Obuf, idx, pos, probs, y);
}
